// ACGMultiHeadAttention_46806553592360
// MI455X (gfx1250) — compile-verified, matrix operand order checked by reading, not run on hardware
//
#include <hip/hip_runtime.h>
#include <hip/hip_bf16.h>

typedef _Float16 half_t;
typedef __attribute__((ext_vector_type(16))) _Float16 v16h;
typedef __attribute__((ext_vector_type(8)))  float    v8f;
typedef __attribute__((ext_vector_type(4)))  unsigned int u32x4;
typedef __attribute__((ext_vector_type(8)))  int i32x8;
typedef __attribute__((ext_vector_type(4)))  int i32x4;

#define DEV __device__ __forceinline__

#if __has_builtin(__builtin_amdgcn_tensor_load_to_lds) && __has_builtin(__builtin_amdgcn_s_wait_tensorcnt)
#define HAVE_TDM 1
#else
#define HAVE_TDM 0
#endif

constexpr int Bn = 64, Sn = 200, Hn = 4, Dn = 256, HDn = 64;
constexpr int SP = 224;                       // S padded to multiple of 32
constexpr int MROW = Bn * Sn;                 // 12800
constexpr float SCALE = 0.125f;               // 1/sqrt(HD)
constexpr size_t SC_BATCH = (size_t)SP * SP;              // per (b,h) score tile (halves)
constexpr size_t SC_CH = (size_t)Bn * Hn * SC_BATCH;      // per score channel (halves)
constexpr int NGATE = 13;                     // 9 (raw) + 4 (raw_c) gate channels

struct CMap13 { int c[13]; };
struct CMap9  { int c[9]; };

// ---------------- WMMA helpers (CDNA5 16x16x32 f16) ----------------

DEV v8f wmma_f16(v16h a, v16h b, v8f c) {
  return __builtin_amdgcn_wmma_f32_16x16x32_f16(false, a, false, b, (short)0, c,
                                                false, false);
}

// Fragment load from a row-major LDS tile (ld halves per row).
// Works for A (rows = M) and for B staged n-major (rows = N).
DEV v16h load_frag16(const half_t* base, int ld, int tile16, int lane) {
  int rm = tile16 * 16 + (lane & 15);
  int kb = (lane >> 4) * 8;
  const half_t* p = base + (size_t)rm * ld;
  union { uint4 u[2]; v16h v; } t;
  t.u[0] = *(const uint4*)(p + kb);
  t.u[1] = *(const uint4*)(p + kb + 16);
  return t.v;
}

#if HAVE_TDM
// On AMDGPU a local->generic addrspace cast keeps the LDS byte offset in the
// low 32 bits of the generic pointer.
DEV unsigned lds_off(const void* p) {
  return (unsigned)(unsigned long long)p;
}

// 2D tile load via Tensor Data Mover: `rows` rows of `rowlen` f16 elements,
// global row stride `stride` elements, packed contiguously into LDS.
// vrows/vcols give the valid tensor extent from the tile start: the TDM
// zero-fills OOB reads, which implements row/col guards for free.
DEV void tdm_load_2d(unsigned ldsByteOff, const half_t* gsrc,
                     unsigned rows, unsigned rowlen, unsigned stride,
                     unsigned vrows, unsigned vcols) {
  unsigned long long ga = (unsigned long long)gsrc;
  u32x4 g0;
  g0[0] = 1u;                                        // count=1, user descriptor
  g0[1] = ldsByteOff;                                // lds_addr (bytes)
  g0[2] = (unsigned)(ga & 0xffffffffull);            // global_addr[31:0]
  g0[3] = (unsigned)((ga >> 32) & 0x01ffffffull) | (2u << 30);  // addr[56:32], type=2
  i32x8 g1;
  g1[0] = 0x00010000;                                // data_size=1 (2 bytes), mask=0
  g1[1] = (int)((vcols & 0xffffu) << 16);            // tensor_dim0[15:0] -> bits 63:48
  g1[2] = (int)(((vcols >> 16) & 0xffffu) | ((vrows & 0xffffu) << 16));
  g1[3] = (int)(((vrows >> 16) & 0xffffu) | ((rowlen & 0xffffu) << 16)); // tile_dim0
  g1[4] = (int)(rows & 0xffffu);                     // tile_dim1 (tile_dim2 = 0)
  g1[5] = (int)stride;                               // tensor_dim0_stride[31:0]
  g1[6] = 0;
  g1[7] = 0;
  i32x4 g2 = {0, 0, 0, 0};
  i32x4 g3 = {0, 0, 0, 0};
#if __clang_major__ >= 23
  i32x8 g4 = {0, 0, 0, 0, 0, 0, 0, 0};
  __builtin_amdgcn_tensor_load_to_lds(g0, g1, g2, g3, g4, 0);
#else
  __builtin_amdgcn_tensor_load_to_lds(g0, g1, g2, g3, 0);
#endif
}
#endif

// ---------------- elementwise kernels ----------------

__global__ void cvt_f16_pad(const float* __restrict__ src, half_t* __restrict__ dst,
                            int srows, int scols, int drows, int dcols) {
  int i = blockIdx.x * 256 + threadIdx.x;
  if (i >= drows * dcols) return;
  int r = i / dcols, c = i - r * dcols;
  float v = (r < srows && c < scols) ? src[(size_t)r * scols + c] : 0.f;
  dst[i] = (half_t)v;
}

// dst[c][r] = src[r][c]  (zero-padded transpose), dst is drows x dcols
__global__ void cvt_f16_tr(const float* __restrict__ src, half_t* __restrict__ dst,
                           int srows, int scols, int drows, int dcols) {
  int i = blockIdx.x * 256 + threadIdx.x;
  if (i >= drows * dcols) return;
  int c = i / dcols, r = i - c * dcols;   // c = original col, r = original row
  float v = (r < srows && c < scols) ? src[(size_t)r * scols + c] : 0.f;
  dst[i] = (half_t)v;
}

// vt[((b*H+h)*HD + d)*SP + j] = V[(b*S+j)*D + h*HD + d]  (j >= S -> 0)
__global__ void transpose_v(const half_t* __restrict__ V, half_t* __restrict__ vt) {
  int i = blockIdx.x * 256 + threadIdx.x;
  if (i >= Bn * Hn * HDn * SP) return;
  int j = i % SP;
  int t = i / SP;
  int d = t % HDn, bh = t / HDn;
  int b = bh >> 2, h = bh & 3;
  half_t v = (half_t)0.f;
  if (j < Sn) v = V[((size_t)b * Sn + j) * Dn + h * HDn + d];
  vt[i] = v;
}

__global__ void zero_f32(float* __restrict__ p, int n) {
  int i = blockIdx.x * 256 + threadIdx.x;
  if (i < n) p[i] = 0.f;
}

// ---------------- 128x128 tile GEMM: C = A(MxK) * WT^T + bias [+resid] ----------------
// WT is the transposed weight (N x K row-major). 8 waves (2x4); wave tile 64x32.
// Double-buffered TDM pipeline: DMA of tile i+1 overlaps WMMA on tile i.

__global__ __launch_bounds__(256)
void gemm128_f16(const half_t* __restrict__ A, const half_t* __restrict__ WT,
                 const float* __restrict__ bias, const float* __restrict__ resid,
                 half_t* __restrict__ out16, float* __restrict__ out32,
                 int M, int K, int N) {
  __shared__ __align__(16) half_t As[2][128 * 32];
  __shared__ __align__(16) half_t Bs[2][128 * 32];  // n-major: [n][k]
  int tid = threadIdx.x, lane = tid & 31, w = tid >> 5;
  int wr = w >> 2, wc = w & 3;
  int tileM = blockIdx.y * 128, tileN = blockIdx.x * 128;
  v8f acc[4][2] = {};
#if HAVE_TDM
  if (w == 0) {
    tdm_load_2d(lds_off(As[0]), &A[(size_t)tileM * K], 128, 32, (unsigned)K, 128, 32);
    tdm_load_2d(lds_off(Bs[0]), &WT[(size_t)tileN * K], 128, 32, (unsigned)K, 128, 32);
  }
#endif
  for (int k0 = 0; k0 < K; k0 += 32) {
    int cur = (k0 >> 5) & 1;
#if HAVE_TDM
    if (w == 0) __builtin_amdgcn_s_wait_tensorcnt(0);
    __syncthreads();                       // tile ready + buf^1 free
    if (w == 0 && k0 + 32 < K) {
      tdm_load_2d(lds_off(As[cur ^ 1]), &A[(size_t)tileM * K + k0 + 32], 128, 32,
                  (unsigned)K, 128, 32);
      tdm_load_2d(lds_off(Bs[cur ^ 1]), &WT[(size_t)tileN * K + k0 + 32], 128, 32,
                  (unsigned)K, 128, 32);
    }
#else
    __syncthreads();
#pragma unroll
    for (int c = tid; c < 512; c += 256) {                       // A: 128 x 32
      int r = c >> 2, cc = (c & 3) * 8;
      *(uint4*)(&As[cur][r * 32 + cc]) =
          *(const uint4*)(&A[(size_t)(tileM + r) * K + k0 + cc]);
    }
#pragma unroll
    for (int c = tid; c < 512; c += 256) {                       // B: 128 x 32 (n-major)
      int r = c >> 2, cc = (c & 3) * 8;
      *(uint4*)(&Bs[cur][r * 32 + cc]) =
          *(const uint4*)(&WT[(size_t)(tileN + r) * K + k0 + cc]);
    }
    __syncthreads();
#endif
    const half_t* Ac = As[cur];
    const half_t* Bc = Bs[cur];
    v16h af[4], bf[2];
#pragma unroll
    for (int i = 0; i < 4; ++i) af[i] = load_frag16(Ac + (size_t)(wr * 64) * 32, 32, i, lane);
#pragma unroll
    for (int j = 0; j < 2; ++j) bf[j] = load_frag16(Bc + (size_t)(wc * 32) * 32, 32, j, lane);
#pragma unroll
    for (int i = 0; i < 4; ++i)
#pragma unroll
      for (int j = 0; j < 2; ++j) acc[i][j] = wmma_f16(af[i], bf[j], acc[i][j]);
  }
#pragma unroll
  for (int i = 0; i < 4; ++i)
#pragma unroll
    for (int j = 0; j < 2; ++j)
#pragma unroll
      for (int r = 0; r < 8; ++r) {
        int row = tileM + wr * 64 + i * 16 + r + ((lane >> 4) << 3);
        int col = tileN + wc * 32 + j * 16 + (lane & 15);
        float v = acc[i][j][r] + bias[col];
        if (resid) v += resid[(size_t)row * N + col];
        if (out16) out16[(size_t)row * N + col] = (half_t)v;
        if (out32) out32[(size_t)row * N + col] = v;
      }
}

// ---------------- batched score GEMM: Sc[bh] = Q(200x64) * K^T, padded 224x224 f16 out ----
// 128 threads = 4 waves (2x2); wave tile 32x32. Row guards via TDM OOB zero-fill.

__global__ __launch_bounds__(128)
void score_gemm(const half_t* __restrict__ Qp, const half_t* __restrict__ Kp,
                half_t* __restrict__ Out) {
  int z = blockIdx.z, b = z >> 2, h = z & 3;
  const half_t* Qb = Qp + (size_t)b * Sn * Dn + h * HDn;
  const half_t* Kb = Kp + (size_t)b * Sn * Dn + h * HDn;
  __shared__ __align__(16) half_t As[2][64 * 32];
  __shared__ __align__(16) half_t Bs[2][64 * 32];   // n-major: [n][k]
  int tid = threadIdx.x, lane = tid & 31, w = tid >> 5;
  int wr = w >> 1, wc = w & 1;
  int tileM = blockIdx.y * 64, tileN = blockIdx.x * 64;
  v8f acc[2][2] = {};
#if HAVE_TDM
  unsigned vA = (unsigned)(Sn - tileM > 0 ? Sn - tileM : 0);
  unsigned vB = (unsigned)(Sn - tileN > 0 ? Sn - tileN : 0);
  if (w == 0) {
    tdm_load_2d(lds_off(As[0]), Qb + (size_t)tileM * Dn, 64, 32, Dn, vA, 32);
    tdm_load_2d(lds_off(Bs[0]), Kb + (size_t)tileN * Dn, 64, 32, Dn, vB, 32);
  }
#endif
  for (int k0 = 0; k0 < HDn; k0 += 32) {
    int cur = (k0 >> 5) & 1;
#if HAVE_TDM
    if (w == 0) __builtin_amdgcn_s_wait_tensorcnt(0);
    __syncthreads();
    if (w == 0 && k0 + 32 < HDn) {
      tdm_load_2d(lds_off(As[cur ^ 1]), Qb + (size_t)tileM * Dn + k0 + 32, 64, 32, Dn, vA, 32);
      tdm_load_2d(lds_off(Bs[cur ^ 1]), Kb + (size_t)tileN * Dn + k0 + 32, 64, 32, Dn, vB, 32);
    }
#else
    __syncthreads();
#pragma unroll
    for (int c = tid; c < 256; c += 128) {
      int r = c >> 2, cc = (c & 3) * 8, gr = tileM + r;
      uint4 v{};
      if (gr < Sn) v = *(const uint4*)(&Qb[(size_t)gr * Dn + k0 + cc]);
      *(uint4*)(&As[cur][r * 32 + cc]) = v;
    }
#pragma unroll
    for (int c = tid; c < 256; c += 128) {
      int n = c >> 2, cc = (c & 3) * 8, gn = tileN + n;
      uint4 v{};
      if (gn < Sn) v = *(const uint4*)(&Kb[(size_t)gn * Dn + k0 + cc]);
      *(uint4*)(&Bs[cur][n * 32 + cc]) = v;
    }
    __syncthreads();
#endif
    v16h af[2], bf[2];
#pragma unroll
    for (int i = 0; i < 2; ++i) af[i] = load_frag16(As[cur] + (size_t)(wr * 32) * 32, 32, i, lane);
#pragma unroll
    for (int j = 0; j < 2; ++j) bf[j] = load_frag16(Bs[cur] + (size_t)(wc * 32) * 32, 32, j, lane);
#pragma unroll
    for (int i = 0; i < 2; ++i)
#pragma unroll
      for (int j = 0; j < 2; ++j) acc[i][j] = wmma_f16(af[i], bf[j], acc[i][j]);
  }
  half_t* Ob = Out + (size_t)z * SC_BATCH;
#pragma unroll
  for (int i = 0; i < 2; ++i)
#pragma unroll
    for (int j = 0; j < 2; ++j)
#pragma unroll
      for (int r = 0; r < 8; ++r) {
        int row = tileM + wr * 32 + i * 16 + r + ((lane >> 4) << 3);
        int col = tileN + wc * 32 + j * 16 + (lane & 15);
        if (row < SP && col < SP) {
          float v = (row < Sn && col < Sn) ? acc[i][j][r] : 0.f;
          Ob[(size_t)row * SP + col] = (half_t)v;
        }
      }
}

// ---------------- gate GEMM: e[bh,g,s] += sum_n relu(fw*(Sc@Wf1)[s,n]+bf1[n]) * Wf2[n] ----
// B = Wf1^T (padded, n-major). One atomic per C row via cross-lane reduction.

__global__ __launch_bounds__(128)
void gate_gemm(const half_t* __restrict__ Scores, const half_t* __restrict__ Wf1t,
               const float* __restrict__ bf1, const float* __restrict__ wf2,
               const float* __restrict__ fw9, const float* __restrict__ fw4,
               float* __restrict__ e, CMap13 cm) {
  int z = blockIdx.z;
  int bh = z / NGATE, g = z - bh * NGATE;
  float fwv = (g < 9) ? fw9[g] : fw4[g - 9];
  const half_t* A = Scores + (size_t)cm.c[g] * SC_CH + (size_t)bh * SC_BATCH;
  __shared__ __align__(16) half_t As[2][64 * 32];
  __shared__ __align__(16) half_t Bs[2][64 * 32];   // n-major: [n][k]
  int tid = threadIdx.x, lane = tid & 31, w = tid >> 5;
  int wr = w >> 1, wc = w & 1;
  int tileM = blockIdx.y * 64, tileN = blockIdx.x * 64;
  v8f acc[2][2] = {};
#if HAVE_TDM
  unsigned vA = (unsigned)(SP - tileM);
  unsigned vB = (unsigned)(SP - tileN);
  if (w == 0) {
    tdm_load_2d(lds_off(As[0]), A + (size_t)tileM * SP, 64, 32, SP, vA, 32);
    tdm_load_2d(lds_off(Bs[0]), Wf1t + (size_t)tileN * SP, 64, 32, SP, vB, 32);
  }
#endif
  for (int k0 = 0; k0 < SP; k0 += 32) {
    int cur = (k0 >> 5) & 1;
#if HAVE_TDM
    if (w == 0) __builtin_amdgcn_s_wait_tensorcnt(0);
    __syncthreads();
    if (w == 0 && k0 + 32 < SP) {
      tdm_load_2d(lds_off(As[cur ^ 1]), A + (size_t)tileM * SP + k0 + 32, 64, 32, SP, vA, 32);
      tdm_load_2d(lds_off(Bs[cur ^ 1]), Wf1t + (size_t)tileN * SP + k0 + 32, 64, 32, SP, vB, 32);
    }
#else
    __syncthreads();
#pragma unroll
    for (int c = tid; c < 256; c += 128) {
      int r = c >> 2, cc = (c & 3) * 8, gr = tileM + r;
      uint4 v{};
      if (gr < SP) v = *(const uint4*)(&A[(size_t)gr * SP + k0 + cc]);
      *(uint4*)(&As[cur][r * 32 + cc]) = v;
    }
#pragma unroll
    for (int c = tid; c < 256; c += 128) {
      int n = c >> 2, cc = (c & 3) * 8, gn = tileN + n;
      uint4 v{};
      if (gn < SP) v = *(const uint4*)(&Wf1t[(size_t)gn * SP + k0 + cc]);
      *(uint4*)(&Bs[cur][n * 32 + cc]) = v;
    }
    __syncthreads();
#endif
    v16h af[2], bf[2];
#pragma unroll
    for (int i = 0; i < 2; ++i) af[i] = load_frag16(As[cur] + (size_t)(wr * 32) * 32, 32, i, lane);
#pragma unroll
    for (int j = 0; j < 2; ++j) bf[j] = load_frag16(Bs[cur] + (size_t)(wc * 32) * 32, 32, j, lane);
#pragma unroll
    for (int i = 0; i < 2; ++i)
#pragma unroll
      for (int j = 0; j < 2; ++j) acc[i][j] = wmma_f16(af[i], bf[j], acc[i][j]);
  }
#pragma unroll
  for (int i = 0; i < 2; ++i)
#pragma unroll
    for (int r = 0; r < 8; ++r) {
      float vsum = 0.f;
#pragma unroll
      for (int j = 0; j < 2; ++j) {
        int col = tileN + wc * 32 + j * 16 + (lane & 15);
        if (col < Sn) {
          float v = fwv * acc[i][j][r] + bf1[col];
          vsum += (v > 0.f) ? v * wf2[col] : 0.f;
        }
      }
#pragma unroll
      for (int m = 1; m < 16; m <<= 1) vsum += __shfl_xor(vsum, m, 32);
      int row = tileM + wr * 32 + i * 16 + r + ((lane >> 4) << 3);
      if ((lane & 15) == 0 && row < SP)
        atomicAdd(&e[(size_t)z * SP + row], vsum);
    }
}

// ---------------- channel softmax + fw-weighted combine + masked row softmax ----------------
// bf2 omitted: adding a constant to all channel logits leaves softmax unchanged.

__global__ __launch_bounds__(256)
void combine_softmax(const half_t* __restrict__ Scores, const float* __restrict__ e,
                     int gbase, int nch, const float* __restrict__ fwp,
                     const float* __restrict__ mask, half_t* __restrict__ P, CMap9 cm) {
  int s = blockIdx.x, bh = blockIdx.y, b = bh >> 2;
  int tid = threadIdx.x;
  half_t* Prow = P + (size_t)bh * SC_BATCH + (size_t)s * SP;
  if (s >= Sn) { if (tid < SP) Prow[tid] = (half_t)0.f; return; }
  __shared__ float wgt[9];
  __shared__ float red[256];
  if (tid == 0) {
    float ev[9], m = -1e30f;
    for (int c = 0; c < nch; ++c) {
      ev[c] = e[((size_t)bh * NGATE + gbase + c) * SP + s];
      m = fmaxf(m, ev[c]);
    }
    float sum = 0.f;
    for (int c = 0; c < nch; ++c) { ev[c] = __expf(ev[c] - m); sum += ev[c]; }
    for (int c = 0; c < nch; ++c) wgt[c] = fwp[c] * ev[c] / sum;
  }
  __syncthreads();
  float logit = -1e30f;
  if (tid < Sn) {
    float accv = 0.f;
    for (int c = 0; c < nch; ++c)
      accv += wgt[c] * (float)Scores[(size_t)cm.c[c] * SC_CH + (size_t)bh * SC_BATCH +
                                     (size_t)s * SP + tid];
    logit = accv * SCALE + mask[(size_t)b * Sn * Sn + (size_t)s * Sn + tid];
  }
  red[tid] = logit; __syncthreads();
  for (int st = 128; st > 0; st >>= 1) {
    if (tid < st) red[tid] = fmaxf(red[tid], red[tid + st]);
    __syncthreads();
  }
  float rmax = red[0]; __syncthreads();
  float ex = (tid < Sn) ? __expf(logit - rmax) : 0.f;
  red[tid] = ex; __syncthreads();
  for (int st = 128; st > 0; st >>= 1) {
    if (tid < st) red[tid] += red[tid + st];
    __syncthreads();
  }
  float rsum = red[0];
  if (tid < SP) Prow[tid] = (half_t)((tid < Sn) ? ex / rsum : 0.f);
}

// ---------------- ctx GEMM: ctx = P(200x224) @ V(224x64) using transposed V copy ----------

__global__ __launch_bounds__(128)
void ctx_gemm(const half_t* __restrict__ P, const half_t* __restrict__ vt,
              half_t* __restrict__ ctx16) {
  int z = blockIdx.z, b = z >> 2, h = z & 3;
  const half_t* Ab = P + (size_t)z * SC_BATCH;
  const half_t* Vt = vt + (size_t)z * HDn * SP;     // [d][j], zero-padded j
  __shared__ __align__(16) half_t As[2][64 * 32];
  __shared__ __align__(16) half_t Bs[2][64 * 32];   // n-major: [d][k=j]
  int tid = threadIdx.x, lane = tid & 31, w = tid >> 5;
  int wr = w >> 1, wc = w & 1;
  int tileM = blockIdx.y * 64;
  v8f acc[2][2] = {};
#if HAVE_TDM
  unsigned vA = (unsigned)(SP - tileM);
  if (w == 0) {
    tdm_load_2d(lds_off(As[0]), Ab + (size_t)tileM * SP, 64, 32, SP, vA, 32);
    tdm_load_2d(lds_off(Bs[0]), Vt, 64, 32, SP, 64, 32);
  }
#endif
  for (int k0 = 0; k0 < SP; k0 += 32) {
    int cur = (k0 >> 5) & 1;
#if HAVE_TDM
    if (w == 0) __builtin_amdgcn_s_wait_tensorcnt(0);
    __syncthreads();
    if (w == 0 && k0 + 32 < SP) {
      tdm_load_2d(lds_off(As[cur ^ 1]), Ab + (size_t)tileM * SP + k0 + 32, 64, 32, SP, vA, 32);
      tdm_load_2d(lds_off(Bs[cur ^ 1]), Vt + k0 + 32, 64, 32, SP, 64, 32);
    }
#else
    __syncthreads();
#pragma unroll
    for (int c = tid; c < 256; c += 128) {
      int r = c >> 2, cc = (c & 3) * 8, gr = tileM + r;
      uint4 v{};
      if (gr < SP) v = *(const uint4*)(&Ab[(size_t)gr * SP + k0 + cc]);
      *(uint4*)(&As[cur][r * 32 + cc]) = v;
    }
#pragma unroll
    for (int c = tid; c < 256; c += 128) {
      int n = c >> 2, cc = (c & 3) * 8;
      *(uint4*)(&Bs[cur][n * 32 + cc]) = *(const uint4*)(&Vt[(size_t)n * SP + k0 + cc]);
    }
    __syncthreads();
#endif
    v16h af[2], bf[2];
#pragma unroll
    for (int i = 0; i < 2; ++i) af[i] = load_frag16(As[cur] + (size_t)(wr * 32) * 32, 32, i, lane);
#pragma unroll
    for (int j = 0; j < 2; ++j) bf[j] = load_frag16(Bs[cur] + (size_t)(wc * 32) * 32, 32, j, lane);
#pragma unroll
    for (int i = 0; i < 2; ++i)
#pragma unroll
      for (int j = 0; j < 2; ++j) acc[i][j] = wmma_f16(af[i], bf[j], acc[i][j]);
  }
#pragma unroll
  for (int i = 0; i < 2; ++i)
#pragma unroll
    for (int j = 0; j < 2; ++j)
#pragma unroll
      for (int r = 0; r < 8; ++r) {
        int row = tileM + wr * 32 + i * 16 + r + ((lane >> 4) << 3);
        int col = wc * 32 + j * 16 + (lane & 15);
        if (row < Sn)
          ctx16[((size_t)b * Sn + row) * Dn + h * HDn + col] = (half_t)acc[i][j][r];
      }
}

// ---------------- layernorm over 256-wide rows ----------------

__global__ __launch_bounds__(256)
void layernorm_k(const float* __restrict__ X, const float* __restrict__ g,
                 const float* __restrict__ bb, float* __restrict__ out) {
  int row = blockIdx.x, tid = threadIdx.x;
  float v = X[(size_t)row * Dn + tid];
  __shared__ float red[256];
  red[tid] = v; __syncthreads();
  for (int st = 128; st > 0; st >>= 1) {
    if (tid < st) red[tid] += red[tid + st];
    __syncthreads();
  }
  float mean = red[0] * (1.f / Dn); __syncthreads();
  float d = v - mean;
  red[tid] = d * d; __syncthreads();
  for (int st = 128; st > 0; st >>= 1) {
    if (tid < st) red[tid] += red[tid + st];
    __syncthreads();
  }
  float var = red[0] * (1.f / Dn);
  out[(size_t)row * Dn + tid] = d * rsqrtf(var + 1e-12f) * g[tid] + bb[tid];
}

// ---------------- host orchestration ----------------

extern "C" void kernel_launch(void* const* d_in, const int* in_sizes, int n_in,
                              void* d_out, int out_size, void* d_ws, size_t ws_size,
                              hipStream_t stream) {
  const float* input = (const float*)d_in[0];
  const float* attrT = (const float*)d_in[1];   // (1,B,S,1,AD) contiguous == (B,S,AD)
  const float* pos   = (const float*)d_in[2];
  const float* hatt  = (const float*)d_in[3];
  const float* mask  = (const float*)d_in[4];
  const float* fw9   = (const float*)d_in[5];
  const float* fw4   = (const float*)d_in[6];
  const float *Wq = (const float*)d_in[7],  *bq = (const float*)d_in[8];
  const float *Wk = (const float*)d_in[9],  *bk = (const float*)d_in[10];
  const float *Wv = (const float*)d_in[11], *bv = (const float*)d_in[12];
  const float *Wqp = (const float*)d_in[13], *bqp = (const float*)d_in[14];
  const float *Wkp = (const float*)d_in[15], *bkp = (const float*)d_in[16];
  const float *Waq = (const float*)d_in[17], *baq = (const float*)d_in[18];
  const float *Wak = (const float*)d_in[19], *bak = (const float*)d_in[20];
  const float *Wav = (const float*)d_in[21], *bav = (const float*)d_in[22];
  const float *Wqic = (const float*)d_in[23], *bqic = (const float*)d_in[24];
  const float *Wqci = (const float*)d_in[25], *bqci = (const float*)d_in[26];
  const float *Wqpc = (const float*)d_in[27], *bqpc = (const float*)d_in[28];
  const float *Wqcp = (const float*)d_in[29], *bqcp = (const float*)d_in[30];
  const float *Wf1 = (const float*)d_in[31], *bf1 = (const float*)d_in[32];
  const float *Wf2 = (const float*)d_in[33];  // bf2 (d_in[34]) softmax-invariant
  const float *Wd  = (const float*)d_in[35], *bd = (const float*)d_in[36];
  const float *ln_g = (const float*)d_in[37], *ln_b = (const float*)d_in[38];
  const float *Wda = (const float*)d_in[39], *bda = (const float*)d_in[40];
  const float *lna_g = (const float*)d_in[41], *lna_b = (const float*)d_in[42];

  char* base = (char*)d_ws;
  size_t off = 0;
  auto alloc = [&](size_t bytes) -> void* {
    void* p = base + off;
    off = (off + bytes + 255) & ~(size_t)255;
    return p;
  };
  const size_t ACTB = (size_t)MROW * Dn * sizeof(half_t);
  const size_t WB   = (size_t)Dn * Dn * sizeof(half_t);

  half_t *x16 = (half_t*)alloc(ACTB), *p16 = (half_t*)alloc(ACTB);
  half_t *a16 = (half_t*)alloc(ACTB), *h16 = (half_t*)alloc(ACTB);
  half_t *wq16 = (half_t*)alloc(WB), *wk16 = (half_t*)alloc(WB), *wv16 = (half_t*)alloc(WB);
  half_t *wqp16 = (half_t*)alloc(WB), *wkp16 = (half_t*)alloc(WB);
  half_t *waq16 = (half_t*)alloc(WB), *wak16 = (half_t*)alloc(WB), *wav16 = (half_t*)alloc(WB);
  half_t *wqic16 = (half_t*)alloc(WB), *wqci16 = (half_t*)alloc(WB);
  half_t *wqpc16 = (half_t*)alloc(WB), *wqcp16 = (half_t*)alloc(WB);
  half_t *wd16 = (half_t*)alloc(WB), *wda16 = (half_t*)alloc(WB);
  half_t *wf1t = (half_t*)alloc((size_t)SP * SP * sizeof(half_t));
  half_t *iq = (half_t*)alloc(ACTB), *ik = (half_t*)alloc(ACTB), *iv = (half_t*)alloc(ACTB);
  half_t *pq = (half_t*)alloc(ACTB), *pk = (half_t*)alloc(ACTB);
  half_t *aq = (half_t*)alloc(ACTB), *ak = (half_t*)alloc(ACTB);
  half_t *haq = (half_t*)alloc(ACTB), *hak = (half_t*)alloc(ACTB), *hav = (half_t*)alloc(ACTB);
  half_t *iqc = (half_t*)alloc(ACTB), *ciq = (half_t*)alloc(ACTB);
  half_t *cqp = (half_t*)alloc(ACTB), *cqp_a = (half_t*)alloc(ACTB), *pqc = (half_t*)alloc(ACTB);
  half_t *scores = (half_t*)alloc(12 * SC_CH * sizeof(half_t));
  float  *ebuf   = (float*)alloc((size_t)Bn * Hn * NGATE * SP * sizeof(float));
  half_t *Pa = (half_t*)alloc((size_t)Bn * Hn * SC_BATCH * sizeof(half_t));
  half_t *Pc = (half_t*)alloc((size_t)Bn * Hn * SC_BATCH * sizeof(half_t));
  half_t *vta = (half_t*)alloc((size_t)Bn * Hn * HDn * SP * sizeof(half_t));
  half_t *vtc = (half_t*)alloc((size_t)Bn * Hn * HDn * SP * sizeof(half_t));
  half_t *ctxa = (half_t*)alloc(ACTB), *ctxc = (half_t*)alloc(ACTB);
  float  *pre1 = (float*)alloc((size_t)MROW * Dn * sizeof(float));
  float  *pre2 = (float*)alloc((size_t)MROW * Dn * sizeof(float));

  auto cvt = [&](const float* s, half_t* d, int sr, int sc, int dr, int dc) {
    int n = dr * dc;
    cvt_f16_pad<<<dim3((n + 255) / 256), dim3(256), 0, stream>>>(s, d, sr, sc, dr, dc);
  };
  auto cvtT = [&](const float* s, half_t* d, int sr, int sc, int dr, int dc) {
    int n = dr * dc;
    cvt_f16_tr<<<dim3((n + 255) / 256), dim3(256), 0, stream>>>(s, d, sr, sc, dr, dc);
  };
  // activations (row-major f16) + weights (transposed f16)
  cvt(input, x16, MROW, Dn, MROW, Dn);
  cvt(pos,   p16, MROW, Dn, MROW, Dn);
  cvt(attrT, a16, MROW, Dn, MROW, Dn);
  cvt(hatt,  h16, MROW, Dn, MROW, Dn);
  cvtT(Wq, wq16, Dn, Dn, Dn, Dn);   cvtT(Wk, wk16, Dn, Dn, Dn, Dn);
  cvtT(Wv, wv16, Dn, Dn, Dn, Dn);   cvtT(Wqp, wqp16, Dn, Dn, Dn, Dn);
  cvtT(Wkp, wkp16, Dn, Dn, Dn, Dn); cvtT(Waq, waq16, Dn, Dn, Dn, Dn);
  cvtT(Wak, wak16, Dn, Dn, Dn, Dn); cvtT(Wav, wav16, Dn, Dn, Dn, Dn);
  cvtT(Wqic, wqic16, Dn, Dn, Dn, Dn); cvtT(Wqci, wqci16, Dn, Dn, Dn, Dn);
  cvtT(Wqpc, wqpc16, Dn, Dn, Dn, Dn); cvtT(Wqcp, wqcp16, Dn, Dn, Dn, Dn);
  cvtT(Wd, wd16, Dn, Dn, Dn, Dn);   cvtT(Wda, wda16, Dn, Dn, Dn, Dn);
  cvtT(Wf1, wf1t, Sn, Sn, SP, SP);  // Wf1^T zero-padded to 224x224

  // 15 projections
  auto proj = [&](const half_t* A, const half_t* W, const float* bias, half_t* o16) {
    gemm128_f16<<<dim3(Dn / 128, MROW / 128), dim3(256), 0, stream>>>(
        A, W, bias, nullptr, o16, nullptr, MROW, Dn, Dn);
  };
  proj(x16, wq16, bq, iq);     proj(x16, wk16, bk, ik);     proj(x16, wv16, bv, iv);
  proj(p16, wqp16, bqp, pq);   proj(p16, wkp16, bkp, pk);
  proj(a16, waq16, baq, aq);   proj(a16, wak16, bak, ak);
  proj(h16, waq16, baq, haq);  proj(h16, wak16, bak, hak);  proj(h16, wav16, bav, hav);
  proj(x16, wqic16, bqic, iqc);
  proj(a16, wqci16, bqci, ciq);
  proj(a16, wqcp16, bqcp, cqp);
  proj(h16, wqcp16, bqcp, cqp_a);
  proj(p16, wqpc16, bqpc, pqc);

  // 12 score channels
  const half_t* qlist[12] = {iq, iqc, iq, ciq, aq, cqp, pq, pqc, pq, haq, cqp_a, pqc};
  const half_t* klist[12] = {ik, ak, pk, ik, ak, pk, ik, ak, pk, hak, pk, hak};
  for (int c = 0; c < 12; ++c)
    score_gemm<<<dim3(4, 4, Bn * Hn), dim3(128), 0, stream>>>(
        qlist[c], klist[c], scores + (size_t)c * SC_CH);

  // gate: e accumulated by atomics
  {
    int n = Bn * Hn * NGATE * SP;
    zero_f32<<<dim3((n + 255) / 256), dim3(256), 0, stream>>>(ebuf, n);
  }
  CMap13 gmap = {{0, 1, 2, 3, 4, 5, 6, 7, 8, 9, 10, 11, 8}};
  gate_gemm<<<dim3(4, 4, Bn * Hn * NGATE), dim3(128), 0, stream>>>(
      scores, wf1t, bf1, Wf2, fw9, fw4, ebuf, gmap);

  // combine + row softmax -> P
  CMap9 cmA = {{0, 1, 2, 3, 4, 5, 6, 7, 8}};
  CMap9 cmC = {{9, 10, 11, 8, 0, 0, 0, 0, 0}};
  combine_softmax<<<dim3(SP, Bn * Hn), dim3(256), 0, stream>>>(
      scores, ebuf, 0, 9, fw9, mask, Pa, cmA);
  combine_softmax<<<dim3(SP, Bn * Hn), dim3(256), 0, stream>>>(
      scores, ebuf, 9, 4, fw4, mask, Pc, cmC);

  // transposed V copies, then ctx = P @ V
  {
    int n = Bn * Hn * HDn * SP;
    transpose_v<<<dim3((n + 255) / 256), dim3(256), 0, stream>>>(iv, vta);
    transpose_v<<<dim3((n + 255) / 256), dim3(256), 0, stream>>>(hav, vtc);
  }
  ctx_gemm<<<dim3(1, 4, Bn * Hn), dim3(128), 0, stream>>>(Pa, vta, ctxa);
  ctx_gemm<<<dim3(1, 4, Bn * Hn), dim3(128), 0, stream>>>(Pc, vtc, ctxc);

  // output projections with residual, then LN into d_out
  gemm128_f16<<<dim3(Dn / 128, MROW / 128), dim3(256), 0, stream>>>(
      ctxa, wd16, bd, input, nullptr, pre1, MROW, Dn, Dn);
  gemm128_f16<<<dim3(Dn / 128, MROW / 128), dim3(256), 0, stream>>>(
      ctxc, wda16, bda, hatt, nullptr, pre2, MROW, Dn, Dn);

  float* out = (float*)d_out;
  layernorm_k<<<dim3(MROW), dim3(256), 0, stream>>>(pre1, ln_g, ln_b, out);
  layernorm_k<<<dim3(MROW), dim3(256), 0, stream>>>(pre2, lna_g, lna_b,
                                                    out + (size_t)MROW * Dn);
}
